// SparseConv2D_32246614458556
// MI455X (gfx1250) — matrix operand order repaired and checked by reading, not compiled, not run on hardware
//
#include <hip/hip_runtime.h>

typedef __attribute__((ext_vector_type(16))) __bf16 v16bf;
typedef __attribute__((ext_vector_type(8)))  float  v8f;

#define BATCH 32
#define CIN   64
#define HH    112
#define WW    112
#define COUT  64
#define KTOT  576            // CIN*9, ordered k = pos*64 + ci  (pos = kh*3+kw)
#define NCHUNK 18            // 576 / 32
#define PIX_PER_IMG (HH * WW)
#define TILE_PIX 64

__device__ __forceinline__ unsigned short f2bf(float f) {
  union { float f; unsigned int u; } v; v.f = f;
  unsigned int u = v.u;
  u += 0x7fffu + ((u >> 16) & 1u);      // round-to-nearest-even
  return (unsigned short)(u >> 16);
}

// ---------------------------------------------------------------------------
// Repack OIHW fp32 weights into bf16, pre-swizzled into the WMMA A-fragment
// layout: offset = ((kc*64 + co)*2 + half)*16 + j   (half0: K{0..7,16..23},
// half1: K{8..15,24..31} per the ISA 16-bit A-matrix table).
// ---------------------------------------------------------------------------
__global__ void repack_w_kernel(const float* __restrict__ w,
                                unsigned short* __restrict__ wp) {
  int idx = blockIdx.x * blockDim.x + threadIdx.x;
  if (idx >= COUT * KTOT) return;
  int kc   = idx >> 11;          // / 2048
  int rem  = idx & 2047;
  int co   = rem >> 5;
  int half = (rem >> 4) & 1;
  int j    = rem & 15;
  int kk   = (j & 7) + half * 8 + ((j >> 3) << 4);   // lane-slot -> K in chunk
  int k    = kc * 32 + kk;
  int pos  = k >> 6;             // kernel tap 0..8
  int ci   = k & 63;
  wp[idx] = f2bf(w[co * 576 + ci * 9 + pos]);
}

// ---------------------------------------------------------------------------
// Implicit-GEMM conv via v_wmma_f32_16x16x32_bf16.
// Block: 256 threads = 8 waves (4 co-tiles x 2 pixel-groups); 64co x 64pix.
// A tile staged with async global->LDS DMA (ASYNCcnt); B tile gathered +
// converted in VGPRs; output stores non-temporal to keep x L2-resident.
// ---------------------------------------------------------------------------
__global__ __launch_bounds__(256)
void conv_wmma_kernel(const float* __restrict__ x,
                      const unsigned short* __restrict__ wp,
                      float* __restrict__ out) {
  __shared__ __align__(128) unsigned short As[2][2048];  // 64co x 32k  (bf16)
  __shared__ __align__(128) unsigned short Bs[2][2048];  // 64px x 32k  (bf16)

  const int tid  = threadIdx.x;
  const int lane = tid & 31;
  const int wv   = tid >> 5;       // 0..7
  const int mi   = wv & 3;         // co tile      (16 rows)
  const int ni   = wv >> 2;        // pixel group  (32 cols)

  const int pix0 = blockIdx.x * TILE_PIX;

  // ---- per-thread staging coordinates (fixed pixel, 8 channels per chunk)
  const int spix = pix0 + (tid & 63);
  const int srow = tid >> 6;                     // 0..3 -> kk = srow*8 + i
  const int sb   = spix / PIX_PER_IMG;
  const int srem = spix - sb * PIX_PER_IMG;
  const int sh   = srem / WW;
  const int sw   = srem - sh * WW;

  // LDS byte address of As base (flat addr low 32 bits == LDS offset)
  const unsigned asBase = (unsigned)(uintptr_t)(&As[0][0]);

  auto stageA = [&](int kc, int buf) {
    // contiguous 4KB chunk in final fragment layout: async DMA, no VGPR trip
    unsigned ldsAddr = asBase + (unsigned)buf * 4096u + (unsigned)tid * 16u;
    unsigned voff    = (unsigned)tid * 16u;
    const unsigned short* src = wp + kc * 2048;          // uniform (SGPR pair)
    asm volatile("global_load_async_to_lds_b128 %0, %1, %2 offset:0"
                 :: "v"(ldsAddr), "v"(voff), "s"(src)
                 : "memory");
  };

  auto stageB = [&](int kc, int buf) {
    const int pos = kc >> 1;                     // one kernel tap per chunk
    const int y   = sh + pos / 3 - 1;
    const int xx  = sw + pos % 3 - 1;
    const bool inb = ((unsigned)y < HH) && ((unsigned)xx < WW);
    const int cibase = (kc & 1) * 32 + srow * 8;
    const float* xp = x + (((sb * CIN + cibase) * HH + y) * WW + xx);
    unsigned short* bs = Bs[buf];
    const int pl = tid & 63;
#pragma unroll
    for (int i = 0; i < 8; ++i) {
      float v = inb ? xp[i * (HH * WW)] : 0.0f;  // zero-pad halo
      int kk   = srow * 8 + i;
      int half = (kk >> 3) & 1;                  // B layout: half0 K{0..7,16..23}, half1 K{8..15,24..31}
      int j    = (kk & 7) + ((kk >> 4) << 3);
      bs[(pl * 2 + half) * 16 + j] = f2bf(v);
    }
  };

  auto prefetchB = [&](int kc) {                 // warm L2/WGP$ for tap kc
    const int pos = kc >> 1;
    const int y   = sh + pos / 3 - 1;
    const int xx  = sw + pos % 3 - 1;
    if (((unsigned)y < HH) && ((unsigned)xx < WW)) {
      const int cibase = (kc & 1) * 32 + srow * 8;
      __builtin_prefetch(x + (((sb * CIN + cibase) * HH + y) * WW + xx), 0, 0);
    }
  };

  // ---- fragment addresses (contiguous 32B per lane; ds_load_b128 pairs)
  const int aoff  = ((mi * 16 + (lane & 15)) * 2 + (lane >> 4)) * 16;
  const int boff0 = ((ni * 32 + (lane & 15)) * 2 + (lane >> 4)) * 16;
  const int boff1 = boff0 + 16 * 32;             // +16 pixels

  v8f acc0 = {}; v8f acc1 = {};

  stageA(0, 0);
  stageB(0, 0);
  asm volatile("s_wait_asynccnt 0" ::: "memory");
  __syncthreads();

  for (int kc = 0; kc < NCHUNK; ++kc) {
    const int buf = kc & 1;
    v16bf a  = *reinterpret_cast<const v16bf*>(&As[buf][aoff]);
    v16bf b0 = *reinterpret_cast<const v16bf*>(&Bs[buf][boff0]);
    v16bf b1 = *reinterpret_cast<const v16bf*>(&Bs[buf][boff1]);

    if (kc + 1 < NCHUNK) {                       // overlap next-tile staging
      stageA(kc + 1, buf ^ 1);
      stageB(kc + 1, buf ^ 1);
    }
    if (kc + 2 < NCHUNK) prefetchB(kc + 2);

    acc0 = __builtin_amdgcn_wmma_f32_16x16x32_bf16(false, a, false, b0,
                                                   (short)0, acc0, false, false);
    acc1 = __builtin_amdgcn_wmma_f32_16x16x32_bf16(false, a, false, b1,
                                                   (short)0, acc1, false, false);

    asm volatile("s_wait_asynccnt 0" ::: "memory");  // async A landed in LDS
    __syncthreads();                                 // + s_wait_dscnt, barrier
  }

  // ---- epilogue: C/D layout M = r + 8*(lane>=16), N = lane&15
  const int hl      = lane >> 4;
  const int co_base = mi * 16;

  const int pg0 = pix0 + ni * 32 + (lane & 15);
  const int b0i = pg0 / PIX_PER_IMG;
  const int r0  = pg0 - b0i * PIX_PER_IMG;
  const int obase0 = (b0i * COUT * HH * WW) + r0;   // + co*HH*WW later

  const int pg1 = pg0 + 16;
  const int b1i = pg1 / PIX_PER_IMG;
  const int r1  = pg1 - b1i * PIX_PER_IMG;
  const int obase1 = (b1i * COUT * HH * WW) + r1;

#pragma unroll
  for (int r = 0; r < 8; ++r) {
    const int co = co_base + r + hl * 8;
    __builtin_nontemporal_store(acc0[r], &out[obase0 + co * (HH * WW)]);
    __builtin_nontemporal_store(acc1[r], &out[obase1 + co * (HH * WW)]);
  }
}

// ---------------------------------------------------------------------------
extern "C" void kernel_launch(void* const* d_in, const int* in_sizes, int n_in,
                              void* d_out, int out_size, void* d_ws, size_t ws_size,
                              hipStream_t stream) {
  (void)in_sizes; (void)n_in; (void)out_size; (void)ws_size;
  const float* x = (const float*)d_in[0];
  const float* w = (const float*)d_in[1];
  float* out = (float*)d_out;
  unsigned short* wp = (unsigned short*)d_ws;   // 73,728 bytes of bf16 weights

  repack_w_kernel<<<(COUT * KTOT + 255) / 256, 256, 0, stream>>>(w, wp);

  const int nblocks = (BATCH * PIX_PER_IMG) / TILE_PIX;  // 6272
  conv_wmma_kernel<<<nblocks, 256, 0, stream>>>(x, wp, out);
}
